// GAT_82935818486070
// MI455X (gfx1250) — matrix operand order repaired and checked by reading, not compiled, not run on hardware
//
#include <hip/hip_runtime.h>
#include <hip/hip_bf16.h>
#include <math.h>

// ---------------------------------------------------------------------------
// 2-layer GAT forward for MI455X (gfx1250, wave32).
// GEMMs via v_wmma_f32_16x16x32_bf16 with 1x4 (1x3) register tiling along N:
// one wave computes a 16x64 (16x48) strip, amortizing each A fragment over
// 4 (3) WMMAs. Edge phase: L2-resident gather + atomic segment softmax.
// ---------------------------------------------------------------------------

typedef __attribute__((ext_vector_type(16))) __bf16 v16bf;
typedef __attribute__((ext_vector_type(8)))  float  v8f;

#define WG 256           // 8 waves per block on wave32
#define GK 256           // K dimension of both GEMMs

// ---- float atomic max via monotonic int/uint bit trick --------------------
__device__ __forceinline__ void atomicMaxF(float* addr, float v) {
    if (v >= 0.0f) atomicMax((int*)addr, __float_as_int(v));
    else           atomicMin((unsigned int*)addr, __float_as_uint(v));
}

// ---- init / conversion helpers --------------------------------------------
__global__ void k_fill(float* __restrict__ p, float v, long n) {
    long i = (long)blockIdx.x * blockDim.x + threadIdx.x;
    if (i < n) p[i] = v;
}

__global__ void k_cvt_bf16(const float* __restrict__ s, __bf16* __restrict__ d, long n) {
    long i = (long)blockIdx.x * blockDim.x + threadIdx.x;
    if (i < n) d[i] = (__bf16)s[i];
}

// W [K x Ncols] row-major (f32) -> WT [Npad x K] row-major (bf16), zero pad.
__global__ void k_transpose_bf16(const float* __restrict__ W, __bf16* __restrict__ WT,
                                 int K, int Ncols, int Npad) {
    long i = (long)blockIdx.x * blockDim.x + threadIdx.x;
    if (i >= (long)Npad * K) return;
    int ncol = (int)(i / K), k = (int)(i % K);
    WT[(size_t)ncol * K + k] = (ncol < Ncols) ? (__bf16)W[(size_t)k * Ncols + ncol]
                                              : (__bf16)0.0f;
}

// ---- WMMA GEMM with register tiling ---------------------------------------
// C[M x ncols] = A[M x GK](bf16) * BT[...]^T.  One wave -> 16 x (16*TN) strip.
// M must be a multiple of 16 (50000 = 3125*16).
union FragAB { v16bf v; uint4 q[2]; };

template<int TN>
__global__ void __launch_bounds__(WG)
k_gemm_bf16(const __bf16* __restrict__ A, const __bf16* __restrict__ BT,
            float* __restrict__ C, int M, int NTG, int ldC, int ncols)
{
    const int wave = (int)((blockIdx.x * (unsigned)WG + threadIdx.x) >> 5);
    const int lane = threadIdx.x & 31;
    const int mtiles = M >> 4;
    if (wave >= mtiles * NTG) return;           // whole-wave exit only
    const int tm  = wave % mtiles;
    const int tg  = wave / mtiles;              // group of TN consecutive N tiles
    const int l15 = lane & 15;
    const int hi  = lane >> 4;                  // 0: lanes 0-15, 1: lanes 16-31

    // A fragment: lane row = tm*16 + l15; lane<16 K {0..7,16..23}, lane>=16 {8..15,24..31}
    const __bf16* arow = A + (size_t)(tm * 16 + l15) * GK;
    // B fragments: lane col = tile*16 + l15; lane<16 K {0..7,16..23}, lane>=16 {8..15,24..31}
    const __bf16* bbase[TN];
    #pragma unroll
    for (int j = 0; j < TN; ++j)
        bbase[j] = BT + (size_t)((tg * TN + j) * 16 + l15) * GK;

    v8f acc[TN];
    #pragma unroll
    for (int j = 0; j < TN; ++j)
        acc[j] = (v8f){0.f, 0.f, 0.f, 0.f, 0.f, 0.f, 0.f, 0.f};

    #pragma unroll
    for (int k0 = 0; k0 < GK; k0 += 32) {
        FragAB a;
        const uint4* ap = (const uint4*)(arow + k0 + hi * 8);
        a.q[0] = ap[0];        // 8 bf16 @ k0 + hi*8
        a.q[1] = ap[2];        // 8 bf16 @ k0 + hi*8 + 16
        #pragma unroll
        for (int j = 0; j < TN; ++j) {
            FragAB b;
            const uint4* bp = (const uint4*)(bbase[j] + k0 + hi * 8);
            b.q[0] = bp[0];    // 8 bf16 @ k0 + hi*8
            b.q[1] = bp[2];
            acc[j] = __builtin_amdgcn_wmma_f32_16x16x32_bf16(
                         false, a.v, false, b.v, (short)0, acc[j], false, false);
        }
    }

    // D layout: VGPR r -> row r (lanes 0-15) / row r+8 (lanes 16-31), col = l15
    #pragma unroll
    for (int j = 0; j < TN; ++j) {
        const int colg = (tg * TN + j) * 16 + l15;
        if (colg < ncols) {
            float* crow = C + (size_t)(tm * 16 + hi * 8) * ldC + colg;
            #pragma unroll
            for (int r = 0; r < 8; ++r) crow[(size_t)r * ldC] = acc[j][r];
        }
    }
}

// ---- layer-1 attention coefficients: one block per node, one wave per head -
__global__ void k_att1(const float* __restrict__ xp,
                       const float* __restrict__ att_src, const float* __restrict__ att_dst,
                       float* __restrict__ a_src, float* __restrict__ a_dst)
{
    const int n = blockIdx.x;
    const int t = threadIdx.x;          // 0..255
    const int h = t >> 5, c = t & 31;   // wave32: one head per wave
    float v = xp[(size_t)n * 256 + t];
    float s = v * att_src[h * 32 + c];
    float d = v * att_dst[h * 32 + c];
    #pragma unroll
    for (int off = 16; off > 0; off >>= 1) {
        s += __shfl_down(s, off);
        d += __shfl_down(d, off);
    }
    if (c == 0) { a_src[(size_t)n * 8 + h] = s; a_dst[(size_t)n * 8 + h] = d; }
}

// ---- layer-2 attention coefficients (H=1, C=40): one thread per node -------
__global__ void k_att2(const float* __restrict__ xp2,
                       const float* __restrict__ as, const float* __restrict__ ad,
                       float* __restrict__ a_src, float* __restrict__ a_dst, long NV)
{
    long n = (long)blockIdx.x * blockDim.x + threadIdx.x;
    if (n >= NV) return;
    float s = 0.f, d = 0.f;
    #pragma unroll 8
    for (int c = 0; c < 40; ++c) {
        float v = xp2[(size_t)n * 40 + c];
        s += v * as[c]; d += v * ad[c];
    }
    a_src[n] = s; a_dst[n] = d;
}

// ---- edge pass A: alpha = leaky_relu(a_src[s]+a_dst[d]); segment max -------
__global__ void k_edge_alpha(const int* __restrict__ src, const int* __restrict__ dst,
                             long E, long NV, int H,
                             const float* __restrict__ a_src, const float* __restrict__ a_dst,
                             float* __restrict__ alpha, float* __restrict__ amax)
{
    long idx = (long)blockIdx.x * blockDim.x + threadIdx.x;
    long tot = (E + NV) * H;
    if (idx >= tot) return;
    long e = idx / H; int h = (int)(idx % H);
    int s, d;
    if (e < E) { s = src[e]; d = dst[e]; } else { s = d = (int)(e - E); }  // self-loops
    float al = a_src[(size_t)s * H + h] + a_dst[(size_t)d * H + h];
    al = (al > 0.f) ? al : 0.2f * al;          // leaky_relu, NEG_SLOPE=0.2
    alpha[idx] = al;
    atomicMaxF(&amax[(size_t)d * H + h], al);
}

// ---- edge pass B: ex = exp(alpha - amax[dst]); segment sum -----------------
__global__ void k_edge_exp(const int* __restrict__ src, const int* __restrict__ dst,
                           long E, long NV, int H,
                           float* __restrict__ alpha, const float* __restrict__ amax,
                           float* __restrict__ denom)
{
    long idx = (long)blockIdx.x * blockDim.x + threadIdx.x;
    long tot = (E + NV) * H;
    if (idx >= tot) return;
    long e = idx / H; int h = (int)(idx % H);
    int d = (e < E) ? dst[e] : (int)(e - E);
    float ex = __expf(alpha[idx] - amax[(size_t)d * H + h]);
    alpha[idx] = ex;                            // overwrite in place
    atomicAdd(&denom[(size_t)d * H + h], ex);
}

// ---- edge pass C: out[dst] += xp[src] * ex/denom[dst] ----------------------
__global__ void k_edge_aggr(const int* __restrict__ src, const int* __restrict__ dst,
                            long E, long NV, int H, int C,
                            const float* __restrict__ ex, const float* __restrict__ denom,
                            const float* __restrict__ xp, float* __restrict__ out)
{
    long idx = (long)blockIdx.x * blockDim.x + threadIdx.x;
    long tot = (E + NV) * H;
    if (idx >= tot) return;
    long e = idx / H; int h = (int)(idx % H);
    int s, d;
    if (e < E) { s = src[e]; d = dst[e]; } else { s = d = (int)(e - E); }
    const int ld = H * C;
    const float* xs = xp  + (size_t)s * ld + h * C;
    float*       od = out + (size_t)d * ld + h * C;
    __builtin_prefetch(xs, 0, 1);               // global_prefetch_b8 into L2/WGP$
    float w = ex[idx] / denom[(size_t)d * H + h];
    for (int c = 0; c < C; ++c) atomicAdd(&od[c], xs[c] * w);
}

// ---- h = bf16(elu(out1 + b1)) ---------------------------------------------
__global__ void k_elu_bias_bf16(const float* __restrict__ out1, const float* __restrict__ b,
                                __bf16* __restrict__ hb, long n, int F)
{
    long i = (long)blockIdx.x * blockDim.x + threadIdx.x;
    if (i >= n) return;
    float v = out1[i] + b[(int)(i % F)];
    v = (v > 0.f) ? v : (__expf(v) - 1.f);      // elu
    hb[i] = (__bf16)v;
}

// ---- final: log_softmax(out2 + b2) over 40 classes -------------------------
__global__ void k_logsoftmax(const float* __restrict__ out2, const float* __restrict__ b2,
                             float* __restrict__ y, long NV)
{
    long n = (long)blockIdx.x * blockDim.x + threadIdx.x;
    if (n >= NV) return;
    float v[40];
    float m = -INFINITY;
    #pragma unroll 8
    for (int c = 0; c < 40; ++c) { v[c] = out2[(size_t)n * 40 + c] + b2[c]; m = fmaxf(m, v[c]); }
    float ssum = 0.f;
    #pragma unroll 8
    for (int c = 0; c < 40; ++c) ssum += __expf(v[c] - m);
    float lse = m + __logf(ssum);
    #pragma unroll 8
    for (int c = 0; c < 40; ++c) y[(size_t)n * 40 + c] = v[c] - lse;
}

// ---------------------------------------------------------------------------
extern "C" void kernel_launch(void* const* d_in, const int* in_sizes, int n_in,
                              void* d_out, int out_size, void* d_ws, size_t ws_size,
                              hipStream_t stream)
{
    const float* x   = (const float*)d_in[0];
    const int*   ei  = (const int*)  d_in[1];
    const float* W1  = (const float*)d_in[2];
    const float* as1 = (const float*)d_in[3];
    const float* ad1 = (const float*)d_in[4];
    const float* b1  = (const float*)d_in[5];
    const float* W2  = (const float*)d_in[6];
    const float* as2 = (const float*)d_in[7];
    const float* ad2 = (const float*)d_in[8];
    const float* b2  = (const float*)d_in[9];

    const long NV = in_sizes[0] / 256;     // 50000 nodes
    const long E  = in_sizes[1] / 2;       // 800000 edges
    const long ET = E + NV;                // + self loops
    const int  H1 = 8, F = 256, F2 = 40, F2P = 48;
    const int* srcI = ei;
    const int* dstI = ei + E;

    // ---- workspace carve-up (256B-aligned regions) ----
    char* ws = (char*)d_ws;
    size_t off = 0;
    auto take = [&](size_t bytes) -> void* {
        void* p = ws + off;
        off = (off + bytes + 255) & ~(size_t)255;
        return p;
    };
    __bf16* xb    = (__bf16*)take((size_t)NV * F * sizeof(__bf16));
    __bf16* W1T   = (__bf16*)take((size_t)F * F * sizeof(__bf16));
    __bf16* W2T   = (__bf16*)take((size_t)F2P * F * sizeof(__bf16));
    float*  xp1   = (float*) take((size_t)NV * F * sizeof(float));
    float*  aS1   = (float*) take((size_t)NV * H1 * sizeof(float));
    float*  aD1   = (float*) take((size_t)NV * H1 * sizeof(float));
    float*  amax1 = (float*) take((size_t)NV * H1 * sizeof(float));
    float*  den1  = (float*) take((size_t)NV * H1 * sizeof(float));
    float*  alp1  = (float*) take((size_t)ET * H1 * sizeof(float));
    float*  out1  = (float*) take((size_t)NV * F * sizeof(float));
    __bf16* hb    = (__bf16*)take((size_t)NV * F * sizeof(__bf16));
    float*  xp2   = (float*) take((size_t)NV * F2 * sizeof(float));
    float*  aS2   = (float*) take((size_t)NV * sizeof(float));
    float*  aD2   = (float*) take((size_t)NV * sizeof(float));
    float*  amax2 = (float*) take((size_t)NV * sizeof(float));
    float*  den2  = (float*) take((size_t)NV * sizeof(float));
    float*  alp2  = (float*) take((size_t)ET * sizeof(float));
    float*  out2  = (float*) take((size_t)NV * F2 * sizeof(float));

    auto nb = [](long n) { return (unsigned)((n + WG - 1) / WG); };

    // ---- init accumulators (d_ws is poisoned by the harness) ----
    k_fill<<<nb(NV * F),  WG, 0, stream>>>(out1,  0.f,       NV * F);
    k_fill<<<nb(NV * H1), WG, 0, stream>>>(amax1, -INFINITY, NV * H1);
    k_fill<<<nb(NV * H1), WG, 0, stream>>>(den1,  0.f,       NV * H1);
    k_fill<<<nb(NV * F2), WG, 0, stream>>>(out2,  0.f,       NV * F2);
    k_fill<<<nb(NV),      WG, 0, stream>>>(amax2, -INFINITY, NV);
    k_fill<<<nb(NV),      WG, 0, stream>>>(den2,  0.f,       NV);

    // ---- precision conversion + weight transpose ----
    k_cvt_bf16<<<nb(NV * F), WG, 0, stream>>>(x, xb, NV * F);
    k_transpose_bf16<<<nb((long)F * F),   WG, 0, stream>>>(W1, W1T, F, F,  F);
    k_transpose_bf16<<<nb((long)F2P * F), WG, 0, stream>>>(W2, W2T, F, F2, F2P);

    // ---- layer 1: xp1 = x @ W1 (WMMA bf16, 16x64 strip per wave) ----
    {
        long waves = (NV / 16) * (F / 64);          // 3125 * 4 = 12500 waves
        k_gemm_bf16<4><<<nb(waves * 32), WG, 0, stream>>>(xb, W1T, xp1, (int)NV, F / 64, F, F);
    }
    k_att1<<<(unsigned)NV, WG, 0, stream>>>(xp1, as1, ad1, aS1, aD1);
    k_edge_alpha<<<nb(ET * H1), WG, 0, stream>>>(srcI, dstI, E, NV, H1, aS1, aD1, alp1, amax1);
    k_edge_exp  <<<nb(ET * H1), WG, 0, stream>>>(srcI, dstI, E, NV, H1, alp1, amax1, den1);
    k_edge_aggr <<<nb(ET * H1), WG, 0, stream>>>(srcI, dstI, E, NV, H1, 32, alp1, den1, xp1, out1);
    k_elu_bias_bf16<<<nb(NV * F), WG, 0, stream>>>(out1, b1, hb, NV * F, F);

    // ---- layer 2: xp2 = h @ W2 (WMMA bf16, 16x48 strip per wave) ----
    {
        long waves = (NV / 16);                     // 3125 waves, TN=3 tiles each
        k_gemm_bf16<3><<<nb(waves * 32), WG, 0, stream>>>(hb, W2T, xp2, (int)NV, 1, F2, F2);
    }
    k_att2<<<nb(NV), WG, 0, stream>>>(xp2, as2, ad2, aS2, aD2, NV);
    k_edge_alpha<<<nb(ET), WG, 0, stream>>>(srcI, dstI, E, NV, 1, aS2, aD2, alp2, amax2);
    k_edge_exp  <<<nb(ET), WG, 0, stream>>>(srcI, dstI, E, NV, 1, alp2, amax2, den2);
    k_edge_aggr <<<nb(ET), WG, 0, stream>>>(srcI, dstI, E, NV, 1, F2, alp2, den2, xp2, out2);

    // ---- log_softmax -> d_out ----
    k_logsoftmax<<<nb(NV), WG, 0, stream>>>(out2, b2, (float*)d_out, NV);
}
